// MSA_6270652252504
// MI455X (gfx1250) — matrix operand order repaired and checked by reading, not compiled, not run on hardware
//
#include <hip/hip_runtime.h>

typedef __attribute__((ext_vector_type(16))) _Float16 v16h;
typedef __attribute__((ext_vector_type(4)))  _Float16 v4h;
typedef __attribute__((ext_vector_type(8)))  float    v8f;
typedef __attribute__((ext_vector_type(4)))  unsigned int u32x4;
typedef __attribute__((ext_vector_type(4)))  int v4i;

#define S_  2048
#define D_  1024
#define H_  16
#define DH_ 64
#define KT_ 64     // keys staged into LDS per block iteration
#define QB_ 256    // queries per block (8 waves x 32 queries)

#if __has_builtin(__builtin_amdgcn_global_load_async_to_lds_b128) && \
    __has_builtin(__builtin_amdgcn_s_wait_asynccnt)
#define ASYNC_LDS 1
#else
#define ASYNC_LDS 0
#endif

__device__ __forceinline__ v8f wmma_f(v16h a, v16h b, v8f c) {
  return __builtin_amdgcn_wmma_f32_16x16x32_f16(
      false, a, false, b, (short)0, c, false, false);
}

// ---- A-operand (M = lane&15; K split at 8 per lane-half, per ISA table) ----
__device__ __forceinline__ v16h load_a_h(const _Float16* p, int ld) {
  int l = threadIdx.x & 31;
  const _Float16* r = p + (size_t)(l & 15) * ld + ((l & 16) ? 8 : 0);
  v16h a;
#pragma unroll
  for (int j = 0; j < 8; ++j) { a[j] = r[j]; a[j + 8] = r[16 + j]; }
  return a;
}
__device__ __forceinline__ v16h load_a_f(const float* __restrict__ p, int ld) {
  int l = threadIdx.x & 31;
  const float* r = p + (size_t)(l & 15) * ld + ((l & 16) ? 8 : 0);
  v16h a;
#pragma unroll
  for (int j = 0; j < 8; ++j) {
    a[j]     = (_Float16)r[j];
    a[j + 8] = (_Float16)r[16 + j];
  }
  return a;
}

// ---- B-operand (N = lane&15; K split at 8 per lane-half, per ISA table) ----
__device__ __forceinline__ v16h load_b_h(const _Float16* p, int ld) {
  int l = threadIdx.x & 31;
  const _Float16* r = p + (size_t)(l & 15) * ld + ((l & 16) ? 8 : 0);
  v16h b;
#pragma unroll
  for (int j = 0; j < 8; ++j) { b[j] = r[j]; b[j + 8] = r[16 + j]; }
  return b;
}
__device__ __forceinline__ v16h load_b_f(const float* __restrict__ p, int ld, float sc) {
  int l = threadIdx.x & 31;
  const float* r = p + (size_t)(l & 15) * ld + ((l & 16) ? 8 : 0);
  v16h b;
#pragma unroll
  for (int j = 0; j < 8; ++j) { b[j] = (_Float16)(r[j] * sc); b[j + 8] = (_Float16)(r[16 + j] * sc); }
  return b;
}

// ============================================================================
// Kernel 1: per-head QKV projection (one wave = 16 rows of one (b,h)).
// Q,K -> f16 [BH,S,DH] (Q pre-scaled by 1/sqrt(DH)); V -> f16 [BH,DH,S] (V^T)
// ============================================================================
__global__ __launch_bounds__(256) void qkv_kernel(
    const float* __restrict__ x,
    const float* __restrict__ Wq, const float* __restrict__ bq,
    const float* __restrict__ Wk, const float* __restrict__ bk,
    const float* __restrict__ Wv, const float* __restrict__ bv,
    _Float16* __restrict__ Qh, _Float16* __restrict__ Kh,
    _Float16* __restrict__ VhT) {
  int wave  = (blockIdx.x * blockDim.x + threadIdx.x) >> 5;
  int stile = wave % (S_ / 16);
  int bh    = wave / (S_ / 16);
  int b = bh / H_, h = bh % H_;
  int l   = threadIdx.x & 31;
  int col = l & 15;
  int hi8 = (l & 16) ? 8 : 0;

  const float* xb = x + ((size_t)(b * S_ + stile * 16)) * D_ + h * DH_;
  v16h a_lo = load_a_f(xb, D_);
  v16h a_hi = load_a_f(xb + 32, D_);

#pragma unroll
  for (int m = 0; m < 3; ++m) {
    const float* W    = (m == 0 ? Wq : (m == 1 ? Wk : Wv)) + (size_t)h * DH_ * DH_;
    const float* bias = (m == 0 ? bq : (m == 1 ? bk : bv)) + h * DH_;
    float sc = (m == 0) ? 0.125f : 1.0f;
#pragma unroll
    for (int et = 0; et < 4; ++et) {
      v16h b_lo = load_b_f(W + (size_t)et * 16 * DH_, DH_, sc);
      v16h b_hi = load_b_f(W + (size_t)et * 16 * DH_ + 32, DH_, sc);
      v8f c = {};
      c = wmma_f(a_lo, b_lo, c);
      c = wmma_f(a_hi, b_hi, c);
      float bb = bias[et * 16 + col] * sc;
      int e = et * 16 + col;
      if (m < 2) {
        _Float16* dst = (m == 0 ? Qh : Kh) + ((size_t)bh * S_ + stile * 16) * DH_ + e;
#pragma unroll
        for (int v = 0; v < 8; ++v)
          dst[(size_t)(v + hi8) * DH_] = (_Float16)(c[v] + bb);
      } else {
        // V^T: rows of consecutive s live in consecutive C VGPRs -> pack v4h
        v4h p0, p1;
#pragma unroll
        for (int v = 0; v < 4; ++v) {
          p0[v] = (_Float16)(c[v] + bb);
          p1[v] = (_Float16)(c[v + 4] + bb);
        }
        _Float16* vp = VhT + ((size_t)bh * DH_ + e) * S_ + stile * 16 + hi8;
        *(v4h*)vp       = p0;
        *(v4h*)(vp + 4) = p1;
      }
    }
  }
}

// ============================================================================
// Kernel 2: flash attention. Block = 8 waves, one (b,h), 256 queries.
// K/V staged in double-buffered LDS (async-to-LDS when available).
// S^T = K*Q^T -> softmax stats in-lane; P^T C-layout == f16 A-operand layout.
// ============================================================================
#if ASYNC_LDS
__device__ __forceinline__ void async_cp16(const void* g, void* l) {
  __builtin_amdgcn_global_load_async_to_lds_b128(
      (__attribute__((address_space(1))) v4i*)g,
      (__attribute__((address_space(3))) v4i*)l, 0, 0);
}
#endif

__global__ __launch_bounds__(256) void attn_kernel(
    const _Float16* __restrict__ Qh, const _Float16* __restrict__ Kh,
    const _Float16* __restrict__ VhT, float* __restrict__ out) {
  __shared__ _Float16 Ks[2][KT_ * DH_];   // [t][d] rows contiguous
  __shared__ _Float16 Vs[2][DH_ * KT_];   // [e][t] rows contiguous (V^T tile)

  int tid = threadIdx.x;
  int wv  = tid >> 5;
  int l   = tid & 31;
  int col = l & 15;
  int hi8 = (l & 16) ? 8 : 0;

  int bh   = blockIdx.x >> 3;            // S_/QB_ == 8
  int qblk = blockIdx.x & 7;
  int b = bh / H_, h = bh % H_;
  int q0 = qblk * QB_ + wv * 32;         // this wave's first query

  const _Float16* Qp = Qh + ((size_t)bh * S_ + q0) * DH_;
  v16h q00 = load_b_h(Qp, DH_);                    // qtile0, d 0..31
  v16h q01 = load_b_h(Qp + 32, DH_);               // qtile0, d 32..63
  v16h q10 = load_b_h(Qp + (size_t)16 * DH_, DH_); // qtile1
  v16h q11 = load_b_h(Qp + (size_t)16 * DH_ + 32, DH_);

  const _Float16* Kbase = Kh  + (size_t)bh * S_ * DH_;
  const _Float16* Vbase = VhT + (size_t)bh * DH_ * S_;

  v8f o[2][4] = {};
  float m_run[2] = {-3.0e30f, -3.0e30f};
  float l_run[2] = {0.0f, 0.0f};

  // ---- stage one KT_ x DH_ K tile + DH_ x KT_ V^T tile into LDS buffer ----
  auto stage = [&](int buf, int t0) {
    _Float16* ks = Ks[buf];
    _Float16* vs = Vs[buf];
    const char* Kg = (const char*)(Kbase + (size_t)t0 * DH_);  // 8KB contiguous
    const char* Vg = (const char*)(Vbase + t0);                // 64 rows x 128B
#pragma unroll
    for (int j = 0; j < 2; ++j) {
      int off = (tid + j * 256) * 16;          // byte offset in 8KB tile
      int e   = off >> 7;                      // V^T row (128B per row)
      int rb  = off & 127;
      const char* gv = Vg + (size_t)e * (S_ * 2) + rb;
#if ASYNC_LDS
      async_cp16(Kg + off, (char*)ks + off);
      async_cp16(gv,       (char*)vs + off);
#else
      u32x4 tk = *(const u32x4*)(Kg + off);
      u32x4 tv = *(const u32x4*)gv;
      *(u32x4*)((char*)ks + off) = tk;
      *(u32x4*)((char*)vs + off) = tv;
#endif
    }
  };

  stage(0, 0);
#if ASYNC_LDS
  __builtin_amdgcn_s_wait_asynccnt(0);
#endif
  __syncthreads();

  int buf = 0;
  for (int t0 = 0; t0 < S_; t0 += KT_, buf ^= 1) {
    if (t0 + KT_ < S_) stage(buf ^ 1, t0 + KT_);

    const _Float16* ks = Ks[buf];
    const _Float16* vs = Vs[buf];

    // ---- GEMM1: S^T tiles for 4 key sub-tiles x 2 query tiles -------------
    v8f st[2][4];
#pragma unroll
    for (int ts = 0; ts < 4; ++ts) {
      v16h ka0 = load_a_h(ks + (size_t)ts * 16 * DH_, DH_);       // d 0..31
      v16h ka1 = load_a_h(ks + (size_t)ts * 16 * DH_ + 32, DH_);  // d 32..63
      v8f c0 = {}, c1 = {};
      c0 = wmma_f(ka0, q00, c0); c0 = wmma_f(ka1, q01, c0);
      c1 = wmma_f(ka0, q10, c1); c1 = wmma_f(ka1, q11, c1);
      st[0][ts] = c0;
      st[1][ts] = c1;
    }

    // ---- online softmax (per-lane; q = col) + pack P as A-operands --------
    v16h pa[2][2];
#pragma unroll
    for (int qt = 0; qt < 2; ++qt) {
      float tm = st[qt][0][0];
#pragma unroll
      for (int ts = 0; ts < 4; ++ts)
#pragma unroll
        for (int v = 0; v < 8; ++v) tm = fmaxf(tm, st[qt][ts][v]);
      tm = fmaxf(tm, __shfl_xor(tm, 16, 32));
      float m_new = fmaxf(m_run[qt], tm);
      float alpha = __expf(m_run[qt] - m_new);
      float rs = 0.0f;
#pragma unroll
      for (int v = 0; v < 8; ++v) {
        float p0 = __expf(st[qt][0][v] - m_new);   // t = v+hi8
        float p1 = __expf(st[qt][1][v] - m_new);   // t = 16+v+hi8
        float p2 = __expf(st[qt][2][v] - m_new);   // t = 32+v+hi8
        float p3 = __expf(st[qt][3][v] - m_new);   // t = 48+v+hi8
        rs += (p0 + p1) + (p2 + p3);
        pa[qt][0][v]     = (_Float16)p0;
        pa[qt][0][v + 8] = (_Float16)p1;
        pa[qt][1][v]     = (_Float16)p2;
        pa[qt][1][v + 8] = (_Float16)p3;
      }
      rs += __shfl_xor(rs, 16, 32);
      l_run[qt] = l_run[qt] * alpha + rs;
      m_run[qt] = m_new;
      // rescale O: O's q index = VGPR v + hi8; alpha lives in lane q
#pragma unroll
      for (int v = 0; v < 8; ++v) {
        float f = __shfl(alpha, v + hi8, 32);
        o[qt][0][v] *= f; o[qt][1][v] *= f;
        o[qt][2][v] *= f; o[qt][3][v] *= f;
      }
    }

    // ---- GEMM2: O += P * V (V^T rows contiguous in LDS) -------------------
#pragma unroll
    for (int et = 0; et < 4; ++et) {
      v16h vb0 = load_b_h(vs + (size_t)et * 16 * KT_, KT_);        // t 0..31
      v16h vb1 = load_b_h(vs + (size_t)et * 16 * KT_ + 32, KT_);   // t 32..63
      o[0][et] = wmma_f(pa[0][0], vb0, o[0][et]);
      o[0][et] = wmma_f(pa[0][1], vb1, o[0][et]);
      o[1][et] = wmma_f(pa[1][0], vb0, o[1][et]);
      o[1][et] = wmma_f(pa[1][1], vb1, o[1][et]);
    }

#if ASYNC_LDS
    __builtin_amdgcn_s_wait_asynccnt(0);
#endif
    __syncthreads();
  }

  // ---- finalize: divide by row sum, store f32 [B,S,D] ----------------------
#pragma unroll
  for (int qt = 0; qt < 2; ++qt) {
    float inv = 1.0f / l_run[qt];
#pragma unroll
    for (int v = 0; v < 8; ++v) {
      float f = __shfl(inv, v + hi8, 32);
      int srow = q0 + qt * 16 + v + hi8;
      float* op = out + ((size_t)(b * S_ + srow)) * D_ + h * DH_ + col;
      op[0]  = o[qt][0][v] * f;
      op[16] = o[qt][1][v] * f;
      op[32] = o[qt][2][v] * f;
      op[48] = o[qt][3][v] * f;
    }
  }
}

extern "C" void kernel_launch(void* const* d_in, const int* in_sizes, int n_in,
                              void* d_out, int out_size, void* d_ws, size_t ws_size,
                              hipStream_t stream) {
  const float* x  = (const float*)d_in[0];
  const float* Wq = (const float*)d_in[1];
  const float* bq = (const float*)d_in[2];
  const float* Wk = (const float*)d_in[3];
  const float* bk = (const float*)d_in[4];
  const float* Wv = (const float*)d_in[5];
  const float* bv = (const float*)d_in[6];
  float* out = (float*)d_out;

  const int B = 2;
  size_t n = (size_t)B * H_ * S_ * DH_;
  _Float16* Qh  = (_Float16*)d_ws;
  _Float16* Kh  = Qh + n;
  _Float16* VhT = Kh + n;

  int waves1 = B * H_ * (S_ / 16);               // 4096
  qkv_kernel<<<waves1 / 8, 256, 0, stream>>>(x, Wq, bq, Wk, bk, Wv, bv, Qh, Kh, VhT);

  int blocks2 = B * H_ * (S_ / QB_);             // 256
  attn_kernel<<<blocks2, 256, 0, stream>>>(Qh, Kh, VhT, out);
}
